// DisorderedCausalSelfAttention_43052752175110
// MI455X (gfx1250) — matrix operand order repaired and checked by reading, not compiled, not run on hardware
//
#include <hip/hip_runtime.h>
#include <hip/hip_bf16.h>
#include <math.h>

// Workspace layout (needs >= 88 MB):
//   [ 0,16MB)  Q   bf16 [B,NH,T,D]   (bQ added)
//   [16,32MB)  K   bf16 [B,NH,T,D]   (bK added)
//   [32,48MB)  Vt  bf16 [B,NH,D,T]   (transposed for PV B-fragments)
//   [48,64MB)  Y   bf16 [B,T,C]
//   [64,70MB)  WtA bf16 [3C,C]       (W_attn transposed, column-major B reads)
//   [70,72MB)  WtP bf16 [C,C]        (W_proj transposed)
//   [72,88MB)  Xb  bf16 [B*T,C]      (x converted once)

typedef __attribute__((ext_vector_type(16))) __bf16 v16bf;
typedef __attribute__((ext_vector_type(8)))  float  v8f;
typedef __attribute__((ext_vector_type(2)))  __bf16 bf16x2;

#define Bn  4
#define Tn  2048
#define Cn  1024
#define NHn 16
#define Dn  64

// 16-bit A-matrix (16x32) K index for (lane, element) per CDNA5 ISA 7.12.2
__device__ __forceinline__ int kmapA(int lane, int e) {
  return ((e < 8) ? e : (e + 8)) + ((lane >= 16) ? 8 : 0);
}
// 16-bit B-matrix (32x16) K index for (lane, element)
__device__ __forceinline__ int kmapB(int lane, int e) {
  return ((e < 8) ? e : (e + 8)) + ((lane >= 16) ? 8 : 0);
}

__device__ __forceinline__ v8f wmma_bf16(v16bf a, v16bf b, v8f c) {
  // (neg_a, A, neg_b, B, c_mod, C, reuse_a, reuse_b)
  return __builtin_amdgcn_wmma_f32_16x16x32_bf16(false, a, false, b, (short)0, c,
                                                 false, false);
}

// load a 16-element bf16 fragment from a per-lane row pointer using kmap pairs
__device__ __forceinline__ v16bf load_fragA(const __bf16* rowp, int lane) {
  v16bf f;
#pragma unroll
  for (int e = 0; e < 16; e += 2) {
    bf16x2 p = *(const bf16x2*)(rowp + kmapA(lane, e));
    f[e] = p[0]; f[e + 1] = p[1];
  }
  return f;
}
__device__ __forceinline__ v16bf load_fragB(const __bf16* rowp, int lane) {
  v16bf f;
#pragma unroll
  for (int e = 0; e < 16; e += 2) {
    bf16x2 p = *(const bf16x2*)(rowp + kmapB(lane, e));
    f[e] = p[0]; f[e + 1] = p[1];
  }
  return f;
}

// ---------------------------------------------------------------------------
// Prep 1: x fp32 -> bf16 (coalesced, 4 elements/thread)
// ---------------------------------------------------------------------------
__global__ __launch_bounds__(256) void cvt_kernel(const float* __restrict__ src,
                                                  __bf16* __restrict__ dst) {
  int i = (blockIdx.x * 256 + threadIdx.x) * 2;   // float2 index
  const float2* s = (const float2*)src;
  bf16x2* d = (bf16x2*)dst;
  float2 a = s[i], b = s[i + 1];
  bf16x2 pa, pb;
  pa[0] = (__bf16)a.x; pa[1] = (__bf16)a.y;
  pb[0] = (__bf16)b.x; pb[1] = (__bf16)b.y;
  d[i] = pa; d[i + 1] = pb;
}

// ---------------------------------------------------------------------------
// Prep 2: W [K][N] fp32 -> Wt [N][K] bf16, 64x64 LDS tiles (coalesced both ways)
// ---------------------------------------------------------------------------
__global__ __launch_bounds__(256) void transpose_kernel(
    const float* __restrict__ W, __bf16* __restrict__ Wt, int K, int N) {
  __shared__ float tile[64][65];
  int tx = threadIdx.x & 63;
  int ty = threadIdx.x >> 6;                     // 0..3
  int ntn = N >> 6;
  int tn = (blockIdx.x % ntn) << 6;
  int tk = (blockIdx.x / ntn) << 6;
#pragma unroll
  for (int r = ty; r < 64; r += 4)
    tile[r][tx] = W[(size_t)(tk + r) * N + tn + tx];
  __syncthreads();
#pragma unroll
  for (int r = ty; r < 64; r += 4)
    Wt[(size_t)(tn + r) * K + tk + tx] = (__bf16)tile[tx][r];
}

// ---------------------------------------------------------------------------
// Kernel 1: qkv = x @ W_attn + b_attn, fused bias add + scatter into Q/K/Vt.
// One wave = one 16x16 output tile; all fragment loads are contiguous bf16
// pairs (A from Xb rows, B from WtA columns). Uniform q/k/v branch on nt.
// ---------------------------------------------------------------------------
__global__ __launch_bounds__(128) void qkv_kernel(
    const __bf16* __restrict__ Xb, const __bf16* __restrict__ WtA,
    const float* __restrict__ bias, const float* __restrict__ bQ,
    const float* __restrict__ bK, __bf16* __restrict__ Qb,
    __bf16* __restrict__ Kb, __bf16* __restrict__ Vt) {
  int lane = threadIdx.x & 31;
  int w    = blockIdx.x * 4 + (threadIdx.x >> 5);
  int mt   = w / 192;
  int nt   = w - mt * 192;
  int nlo  = lane & 15;
  int hi   = lane >> 4;
  int col  = nt * 16 + nlo;                      // output column (0..3071)
  const __bf16* xr = Xb  + (size_t)(mt * 16 + nlo) * Cn;
  const __bf16* wr = WtA + (size_t)col * Cn;

  v8f acc = {};
#pragma unroll 2
  for (int k0 = 0; k0 < Cn; k0 += 32) {
    v16bf af = load_fragA(xr + k0, lane);
    v16bf bf = load_fragB(wr + k0, lane);
    acc = wmma_bf16(af, bf, acc);
  }

  // wave-uniform segment select (16-wide tiles never straddle a segment)
  int seg  = nt >> 6;                            // 0=q 1=k 2=v
  int nth  = nt & 63;
  int h    = nth >> 2;
  int d    = ((nth & 3) << 4) + nlo;
  int cseg = (nth << 4) + nlo;                   // column within segment
  float badd = bias[col];
  if (seg == 0) {
    badd += bQ[cseg];
#pragma unroll
    for (int r = 0; r < 8; ++r) {
      int bt = mt * 16 + r + hi * 8;
      int b = bt >> 11, t = bt & (Tn - 1);
      Qb[(((size_t)b * NHn + h) * Tn + t) * Dn + d] = (__bf16)(acc[r] + badd);
    }
  } else if (seg == 1) {
    badd += bK[cseg];
#pragma unroll
    for (int r = 0; r < 8; ++r) {
      int bt = mt * 16 + r + hi * 8;
      int b = bt >> 11, t = bt & (Tn - 1);
      Kb[(((size_t)b * NHn + h) * Tn + t) * Dn + d] = (__bf16)(acc[r] + badd);
    }
  } else {
#pragma unroll
    for (int r = 0; r < 8; ++r) {
      int bt = mt * 16 + r + hi * 8;
      int b = bt >> 11, t = bt & (Tn - 1);
      Vt[(((size_t)b * NHn + h) * Dn + d) * Tn + t] = (__bf16)(acc[r] + badd);
    }
  }
}

// ---------------------------------------------------------------------------
// Kernel 2: flash attention. One wave = one 16-row query tile of one (b,h).
// Streams 32-key steps: S = Q K^T (4 WMMA), online softmax (f32, shfl_xor
// row reductions inside each 16-lane half), P restriped via per-wave LDS
// tile, O += P V (4 WMMA). Prefetch next K/V tile (global_prefetch_b8).
// ---------------------------------------------------------------------------
__global__ __launch_bounds__(256) void attn_kernel(
    const __bf16* __restrict__ Qb, const __bf16* __restrict__ Kb,
    const __bf16* __restrict__ Vt, __bf16* __restrict__ Yb) {
  __shared__ __bf16 plds[8][16 * 32];
  int lane  = threadIdx.x & 31;
  int wv    = threadIdx.x >> 5;
  int w     = blockIdx.x * 8 + wv;
  int qt    = w & 127;
  int bh    = w >> 7;
  int nlo   = lane & 15;
  int hi    = lane >> 4;
  int qbase = qt * 16;

  const __bf16* Qh = Qb + (size_t)bh * Tn * Dn;
  const __bf16* Kh = Kb + (size_t)bh * Tn * Dn;
  const __bf16* Vh = Vt + (size_t)bh * Dn * Tn;

  // Q A-fragments, two K=32 chunks over D=64 (held for the whole sweep)
  const __bf16* qr = Qh + (size_t)(qbase + nlo) * Dn;
  v16bf qa0 = load_fragA(qr, lane);
  v16bf qa1 = load_fragA(qr + 32, lane);

  v8f o0 = {}, o1 = {}, o2 = {}, o3 = {};
  float mrow[8], lrow[8];
#pragma unroll
  for (int r = 0; r < 8; ++r) { mrow[r] = -3.0e38f; lrow[r] = 0.0f; }

  int nk = (qbase + 47) >> 5;                    // 32-key steps incl. diagonal
  for (int it = 0; it < nk; ++it) {
    int kb = it * 32;
    if (it + 1 < nk) {                           // wave-uniform prefetch
      __builtin_prefetch(Kh + (size_t)(kb + 32 + nlo) * Dn, 0, 3);
      __builtin_prefetch(Vh + (size_t)nlo * Tn + kb + 32, 0, 3);
    }

    const __bf16* kr0 = Kh + (size_t)(kb + nlo) * Dn;
    const __bf16* kr1 = Kh + (size_t)(kb + 16 + nlo) * Dn;
    v16bf kf00 = load_fragB(kr0, lane);
    v16bf kf01 = load_fragB(kr0 + 32, lane);
    v16bf kf10 = load_fragB(kr1, lane);
    v16bf kf11 = load_fragB(kr1 + 32, lane);

    v8f s0 = {}, s1 = {};
    s0 = wmma_bf16(qa0, kf00, s0);
    s0 = wmma_bf16(qa1, kf01, s0);
    s1 = wmma_bf16(qa0, kf10, s1);
    s1 = wmma_bf16(qa1, kf11, s1);

    // scale + causal mask + online softmax (rows live in 16-lane halves)
#pragma unroll
    for (int r = 0; r < 8; ++r) {
      int q = qbase + r + hi * 8;
      float a0 = s0[r] * 0.125f;                 // 1/sqrt(64)
      float a1 = s1[r] * 0.125f;
      if (kb + nlo > q)      a0 = -3.0e38f;
      if (kb + 16 + nlo > q) a1 = -3.0e38f;
      float mx = fmaxf(a0, a1);
      mx = fmaxf(mx, __shfl_xor(mx, 1, 32));
      mx = fmaxf(mx, __shfl_xor(mx, 2, 32));
      mx = fmaxf(mx, __shfl_xor(mx, 4, 32));
      mx = fmaxf(mx, __shfl_xor(mx, 8, 32));
      float mn = fmaxf(mrow[r], mx);
      float al = __expf(mrow[r] - mn);
      mrow[r]  = mn;
      float p0 = __expf(a0 - mn);
      float p1 = __expf(a1 - mn);
      s0[r] = p0; s1[r] = p1;
      float rs = p0 + p1;
      rs += __shfl_xor(rs, 1, 32);
      rs += __shfl_xor(rs, 2, 32);
      rs += __shfl_xor(rs, 4, 32);
      rs += __shfl_xor(rs, 8, 32);
      lrow[r] = lrow[r] * al + rs;
      o0[r] *= al; o1[r] *= al; o2[r] *= al; o3[r] *= al;
    }

    // restripe P: C-layout f32 -> LDS (m,n) bf16 -> A-layout fragment.
    // Single-wave tile: LDS ops are in-order within a wave, no barrier needed.
    __bf16* pl = plds[wv];
#pragma unroll
    for (int r = 0; r < 8; ++r) {
      int m = r + hi * 8;
      pl[m * 32 + nlo]      = (__bf16)s0[r];
      pl[m * 32 + 16 + nlo] = (__bf16)s1[r];
    }
    v16bf pa = load_fragA(pl + nlo * 32, lane);

    // O += P V  (V transposed: contiguous key pairs per lane)
#define PV_STEP(NT, OACC)                                                  \
    {                                                                      \
      v16bf vf = load_fragB(Vh + (size_t)((NT) * 16 + nlo) * Tn + kb, lane); \
      OACC = wmma_bf16(pa, vf, OACC);                                      \
    }
    PV_STEP(0, o0) PV_STEP(1, o1) PV_STEP(2, o2) PV_STEP(3, o3)
#undef PV_STEP
  }

  // normalize and write y in [B,T,C] bf16
  int bq = bh >> 4, h = bh & 15;
#pragma unroll
  for (int r = 0; r < 8; ++r) {
    float inv = 1.0f / lrow[r];
    int t = qbase + r + hi * 8;
    size_t base = ((size_t)bq * Tn + t) * Cn + (size_t)h * Dn;
    Yb[base + 0 * 16 + nlo] = (__bf16)(o0[r] * inv);
    Yb[base + 1 * 16 + nlo] = (__bf16)(o1[r] * inv);
    Yb[base + 2 * 16 + nlo] = (__bf16)(o2[r] * inv);
    Yb[base + 3 * 16 + nlo] = (__bf16)(o3[r] * inv);
  }
}

// ---------------------------------------------------------------------------
// Kernel 3: out = y @ W_proj + b_proj  (fp32 output). M=8192, N=1024.
// ---------------------------------------------------------------------------
__global__ __launch_bounds__(128) void proj_kernel(
    const __bf16* __restrict__ Yb, const __bf16* __restrict__ WtP,
    const float* __restrict__ bias, float* __restrict__ out) {
  int lane = threadIdx.x & 31;
  int w    = blockIdx.x * 4 + (threadIdx.x >> 5);
  int mt   = w >> 6;
  int nt   = w & 63;
  int nlo  = lane & 15;
  int hi   = lane >> 4;
  int col  = nt * 16 + nlo;
  const __bf16* ar = Yb  + (size_t)(mt * 16 + nlo) * Cn;
  const __bf16* wr = WtP + (size_t)col * Cn;

  v8f acc = {};
#pragma unroll 2
  for (int k0 = 0; k0 < Cn; k0 += 32) {
    v16bf af = load_fragA(ar + k0, lane);
    v16bf bf = load_fragB(wr + k0, lane);
    acc = wmma_bf16(af, bf, acc);
  }
  float bb = bias[col];
#pragma unroll
  for (int r = 0; r < 8; ++r) {
    int m = r + hi * 8;
    out[(size_t)(mt * 16 + m) * Cn + col] = acc[r] + bb;
  }
}

// Reference returns a tuple (out, bQ): append bQ after out.
__global__ void tail_kernel(const float* __restrict__ bQ,
                            float* __restrict__ dst) {
  dst[threadIdx.x] = bQ[threadIdx.x];
}

extern "C" void kernel_launch(void* const* d_in, const int* in_sizes, int n_in,
                              void* d_out, int out_size, void* d_ws,
                              size_t ws_size, hipStream_t stream) {
  const float* x      = (const float*)d_in[0];
  const float* W_attn = (const float*)d_in[1];
  const float* b_attn = (const float*)d_in[2];
  const float* W_proj = (const float*)d_in[3];
  const float* b_proj = (const float*)d_in[4];
  const float* bQ     = (const float*)d_in[5];
  const float* bK     = (const float*)d_in[6];
  float* out = (float*)d_out;

  char* ws = (char*)d_ws;
  const size_t MB = (size_t)1024 * 1024;
  __bf16* Qb  = (__bf16*)(ws + 0 * MB);
  __bf16* Kb  = (__bf16*)(ws + 16 * MB);
  __bf16* Vt  = (__bf16*)(ws + 32 * MB);
  __bf16* Yb  = (__bf16*)(ws + 48 * MB);
  __bf16* WtA = (__bf16*)(ws + 64 * MB);
  __bf16* WtP = (__bf16*)(ws + 70 * MB);
  __bf16* Xb  = (__bf16*)(ws + 72 * MB);

  // prep: x -> bf16 ; W_attn/W_proj -> bf16 transposed (column-major B reads)
  cvt_kernel<<<8192, 256, 0, stream>>>(x, Xb);                 // 8M elements
  transpose_kernel<<<768, 256, 0, stream>>>(W_attn, WtA, Cn, 3 * Cn);
  transpose_kernel<<<256, 256, 0, stream>>>(W_proj, WtP, Cn, Cn);

  // 512 * 192 wave-tiles, 4 waves (128 thr) per block
  qkv_kernel<<<24576, 128, 0, stream>>>(Xb, WtA, b_attn, bQ, bK, Qb, Kb, Vt);
  // 64 (b,h) * 128 q-tiles = 8192 waves, 8 waves per block
  attn_kernel<<<1024, 256, 0, stream>>>(Qb, Kb, Vt, Yb);
  // 512 * 64 wave-tiles, 4 waves per block
  proj_kernel<<<8192, 128, 0, stream>>>(Yb, WtP, b_proj, out);
  tail_kernel<<<1, 1024, 0, stream>>>(bQ, out + (size_t)Bn * Tn * Cn);
}
